// MambaScan_63178968924453
// MI455X (gfx1250) — matrix operand order repaired and checked by reading, not compiled, not run on hardware
//
#include <hip/hip_runtime.h>

// ---------------------------------------------------------------------------
// MambaScan for MI455X (gfx1250, wave32, WMMA)
//   1) split fp32 -> bf16 hi/lo (u and the 3 weight matrices)
//   2) fused GEMM (u*WB^T, u*Wg^T) via v_wmma_f32_16x16x32_bf16 (bf16x3 split)
//      + fused sigmoid/exp epilogue producing A_gated, b_mod
//   3) exact sequential scan per (batch, d) channel (matches reference fp32)
//      writing x split to bf16 hi/lo
//   4) GEMM x*WC^T + bC -> out (fp32)
// ---------------------------------------------------------------------------

typedef __attribute__((ext_vector_type(16))) __bf16 v16bf;
typedef __attribute__((ext_vector_type(8)))  __bf16 v8bf;
typedef __attribute__((ext_vector_type(8)))  float  v8f;

#define DDIM 1024
#define LSEQ 4096
#define BATCH 4
#define NTOK (BATCH * LSEQ)      // 16384 tokens
#define EPSV 1e-6f

union V16U { v16bf v; v8bf h[2]; };

__device__ __forceinline__ v8f wmma_bf16x3(v16bf ah, v16bf al, v16bf bh, v16bf bl, v8f c) {
    // hi*hi + hi*lo + lo*hi  ~= fp32 product (drop lo*lo, ~2^-16 relative)
    c = __builtin_amdgcn_wmma_f32_16x16x32_bf16(false, ah, false, bh, (short)0, c, false, false);
    c = __builtin_amdgcn_wmma_f32_16x16x32_bf16(false, ah, false, bl, (short)0, c, false, false);
    c = __builtin_amdgcn_wmma_f32_16x16x32_bf16(false, al, false, bh, (short)0, c, false, false);
    return c;
}

// A-fragment (16x32 bf16) load: per-lane elements j=0..15 map
// to K = j + (lane>=16)*16  -> two adjacent 16B chunks (matches B-fragment).
__device__ __forceinline__ v16bf load_afrag(const __bf16* __restrict__ base,
                                            long row, int k0, int hi) {
    const __bf16* p = base + row * DDIM + k0 + hi * 16;
    V16U a;
    a.h[0] = *(const v8bf*)(p);
    a.h[1] = *(const v8bf*)(p + 8);
    return a.v;
}

// B-fragment (32x16 bf16): B[k][n] = W[n][k]; per-lane j=0..15 maps to
// K = k0 + (lane>=16)*16 + j -> one contiguous 32B load along W's row.
__device__ __forceinline__ v16bf load_bfrag(const __bf16* __restrict__ w,
                                            long ncol, int k0, int hi) {
    return *(const v16bf*)(w + ncol * DDIM + k0 + hi * 16);
}

__device__ __forceinline__ float sigmoidf(float x) { return 1.0f / (1.0f + expf(-x)); }

// --------------------------- split fp32 -> bf16 hi/lo -----------------------
__global__ __launch_bounds__(256) void split_kernel(const float* __restrict__ src,
                                                    __bf16* __restrict__ hi,
                                                    __bf16* __restrict__ lo, int n) {
    int i = blockIdx.x * blockDim.x + threadIdx.x;
    if (i < n) {
        float v = src[i];
        __bf16 h = (__bf16)v;
        hi[i] = h;
        lo[i] = (__bf16)(v - (float)h);
    }
}

// -------------------- fused GEMM for b = u*WB^T, g = u*Wg^T ------------------
// block = 8 waves; block tile 128(M) x 64(N); each wave: 16 rows x 4 n-subtiles.
__global__ __launch_bounds__(256) void gemm_bg_kernel(
    const __bf16* __restrict__ uh, const __bf16* __restrict__ ul,
    const __bf16* __restrict__ wbh, const __bf16* __restrict__ wbl,
    const __bf16* __restrict__ wgh, const __bf16* __restrict__ wgl,
    const float* __restrict__ biasB, const float* __restrict__ biasG,
    const float* __restrict__ u, const float* __restrict__ lam,
    float* __restrict__ Ag, float* __restrict__ bmod) {
    const int wave = threadIdx.x >> 5;
    const int lane = threadIdx.x & 31;
    const int hi = lane >> 4;
    const int ln = lane & 15;
    const long m0 = (long)blockIdx.x * 128 + wave * 16;
    const long n0 = (long)blockIdx.y * 64;
    const long mrow = m0 + ln;

    v8f accB[4], accG[4];
#pragma unroll
    for (int t = 0; t < 4; ++t) { accB[t] = 0.0f; accG[t] = 0.0f; }

    for (int k0 = 0; k0 < DDIM; k0 += 32) {
        v16bf ah = load_afrag(uh, mrow, k0, hi);
        v16bf al = load_afrag(ul, mrow, k0, hi);
#pragma unroll
        for (int t = 0; t < 4; ++t) {
            long nc = n0 + t * 16 + ln;
            v16bf bh = load_bfrag(wbh, nc, k0, hi);
            v16bf bl = load_bfrag(wbl, nc, k0, hi);
            accB[t] = wmma_bf16x3(ah, al, bh, bl, accB[t]);
            v16bf gh = load_bfrag(wgh, nc, k0, hi);
            v16bf gl = load_bfrag(wgl, nc, k0, hi);
            accG[t] = wmma_bf16x3(ah, al, gh, gl, accG[t]);
        }
    }

    // C/D layout: VGPR r holds row (r + hi*8), col = lane&15
#pragma unroll
    for (int t = 0; t < 4; ++t) {
        long e = n0 + t * 16 + ln;
        float bB = biasB[e], bG = biasG[e], lm = lam[e];
#pragma unroll
        for (int r = 0; r < 8; ++r) {
            long row = m0 + r + hi * 8;
            float blin = accB[t][r] + bB;
            float glin = accG[t][r] + bG;
            float g = sigmoidf(glin);
            float uv = u[row * DDIM + e];
            float delta = sigmoidf(uv);
            float Aval = expf(-delta * lm);
            Ag[row * DDIM + e]   = g * Aval;
            bmod[row * DDIM + e] = (1.0f - g) * blin;
        }
    }
}

// ------------------------- exact sequential scan -----------------------------
// one thread per (batch, d) channel; matches reference cumprod/cumsum in fp32.
__global__ __launch_bounds__(256) void scan_kernel(const float* __restrict__ Ag,
                                                   const float* __restrict__ bmod,
                                                   __bf16* __restrict__ xhi,
                                                   __bf16* __restrict__ xlo) {
    int ch = blockIdx.x * blockDim.x + threadIdx.x;  // 0..4095
    int bi = ch >> 10;
    int d  = ch & (DDIM - 1);
    long base = (long)bi * LSEQ * DDIM + d;
    float P = 1.0f, S = 0.0f;
    for (int l = 0; l < LSEQ; ++l) {
        long idx = base + (long)l * DDIM;
        float a = Ag[idx];
        float b = bmod[idx];
        P = P * (a + EPSV);
        S = S + b / (P + EPSV);
        float x = P * S;
        __bf16 h = (__bf16)x;
        xhi[idx] = h;
        xlo[idx] = (__bf16)(x - (float)h);
    }
}

// --------------------------- out = x*WC^T + bC -------------------------------
__global__ __launch_bounds__(256) void gemm_c_kernel(
    const __bf16* __restrict__ xh, const __bf16* __restrict__ xl,
    const __bf16* __restrict__ wch, const __bf16* __restrict__ wcl,
    const float* __restrict__ biasC, float* __restrict__ out) {
    const int wave = threadIdx.x >> 5;
    const int lane = threadIdx.x & 31;
    const int hi = lane >> 4;
    const int ln = lane & 15;
    const long m0 = (long)blockIdx.x * 128 + wave * 16;
    const long n0 = (long)blockIdx.y * 64;
    const long mrow = m0 + ln;

    v8f acc[4];
#pragma unroll
    for (int t = 0; t < 4; ++t) acc[t] = 0.0f;

    for (int k0 = 0; k0 < DDIM; k0 += 32) {
        v16bf ah = load_afrag(xh, mrow, k0, hi);
        v16bf al = load_afrag(xl, mrow, k0, hi);
#pragma unroll
        for (int t = 0; t < 4; ++t) {
            long nc = n0 + t * 16 + ln;
            v16bf bh = load_bfrag(wch, nc, k0, hi);
            v16bf bl = load_bfrag(wcl, nc, k0, hi);
            acc[t] = wmma_bf16x3(ah, al, bh, bl, acc[t]);
        }
    }

#pragma unroll
    for (int t = 0; t < 4; ++t) {
        long e = n0 + t * 16 + ln;
        float bC = biasC[e];
#pragma unroll
        for (int r = 0; r < 8; ++r) {
            long row = m0 + r + hi * 8;
            out[row * DDIM + e] = acc[t][r] + bC;
        }
    }
}

// ---------------------------------------------------------------------------
extern "C" void kernel_launch(void* const* d_in, const int* in_sizes, int n_in,
                              void* d_out, int out_size, void* d_ws, size_t ws_size,
                              hipStream_t stream) {
    // setup_inputs order: u, WB, bB, WC, bC, Wg, bg, lam
    const float* u   = (const float*)d_in[0];
    const float* WB  = (const float*)d_in[1];
    const float* bB  = (const float*)d_in[2];
    const float* WC  = (const float*)d_in[3];
    const float* bC  = (const float*)d_in[4];
    const float* Wg  = (const float*)d_in[5];
    const float* bg  = (const float*)d_in[6];
    const float* lam = (const float*)d_in[7];
    float* out = (float*)d_out;

    const size_t nU = (size_t)NTOK * DDIM;   // 16,777,216
    const size_t nW = (size_t)DDIM * DDIM;   //  1,048,576

    char* p = (char*)d_ws;
    __bf16* u_hi  = (__bf16*)p; p += nU * 2;
    __bf16* u_lo  = (__bf16*)p; p += nU * 2;
    __bf16* WB_hi = (__bf16*)p; p += nW * 2;
    __bf16* WB_lo = (__bf16*)p; p += nW * 2;
    __bf16* Wg_hi = (__bf16*)p; p += nW * 2;
    __bf16* Wg_lo = (__bf16*)p; p += nW * 2;
    __bf16* WC_hi = (__bf16*)p; p += nW * 2;
    __bf16* WC_lo = (__bf16*)p; p += nW * 2;
    float*  AgBuf = (float*)p;  p += nU * 4;
    float*  bmBuf = (float*)p;  p += nU * 4;
    // x hi/lo reuse the u hi/lo buffers (u split no longer needed after gemm_bg)
    __bf16* x_hi = u_hi;
    __bf16* x_lo = u_lo;

    // 1) fp32 -> bf16 hi/lo splits
    split_kernel<<<(int)(nU / 256), 256, 0, stream>>>(u,  u_hi,  u_lo,  (int)nU);
    split_kernel<<<(int)(nW / 256), 256, 0, stream>>>(WB, WB_hi, WB_lo, (int)nW);
    split_kernel<<<(int)(nW / 256), 256, 0, stream>>>(Wg, Wg_hi, Wg_lo, (int)nW);
    split_kernel<<<(int)(nW / 256), 256, 0, stream>>>(WC, WC_hi, WC_lo, (int)nW);

    // 2) fused dual GEMM + gating epilogue
    dim3 ggrid(NTOK / 128, DDIM / 64);
    gemm_bg_kernel<<<ggrid, 256, 0, stream>>>(u_hi, u_lo, WB_hi, WB_lo, Wg_hi, Wg_lo,
                                              bB, bg, u, lam, AgBuf, bmBuf);

    // 3) exact sequential scan (4096 channels)
    scan_kernel<<<(BATCH * DDIM) / 256, 256, 0, stream>>>(AgBuf, bmBuf, x_hi, x_lo);

    // 4) output GEMM
    gemm_c_kernel<<<ggrid, 256, 0, stream>>>(x_hi, x_lo, WC_hi, WC_lo, bC, out);
}
